// MoleculeGCN_75935021793382
// MI455X (gfx1250) — matrix operand order repaired and checked by reading, not compiled, not run on hardware
//
#include <hip/hip_runtime.h>

// ============================================================================
// MoleculeGCN for MI455X (gfx1250): all GEMMs via v_wmma_f32_16x16x32_bf16,
// f32 accumulate; X-tile gathers via global_load_async_to_lds_b128 (ASYNCcnt);
// segment-sums via L2-resident global_atomic_add_f32.
// Pipeline: zero-agg -> prep(bf16 transposed weights + folded biases)
//        -> embed nodes/edges (WMMA, K padded to 32)
//        -> edge MLP (K=384 fused concat in LDS, 2-layer, atomic scatter)
//        -> node MLP (same shape) -> global MLP (WMMA + tiny VALU tail).
// ============================================================================

#define DEV __device__ __forceinline__

typedef __attribute__((ext_vector_type(16))) __bf16 v16bf;
typedef __attribute__((ext_vector_type(8)))  float  v8f;

union Frag {
  uint4 q[2];
  v16bf b;
};

DEV unsigned short f2bf(float f) {
  unsigned int u = __float_as_uint(f);
  u += 0x7FFFu + ((u >> 16) & 1u);   // round-to-nearest-even
  return (unsigned short)(u >> 16);
}
DEV float bf2f(unsigned short h) { return __uint_as_float((unsigned)h << 16); }

DEV uint4 pack8(const float* p) {    // 8 f32 -> 8 bf16 (p must be 32B aligned)
  float4 a = *(const float4*)p;
  float4 b = *(const float4*)(p + 4);
  uint4 r;
  r.x = (unsigned)f2bf(a.x) | ((unsigned)f2bf(a.y) << 16);
  r.y = (unsigned)f2bf(a.z) | ((unsigned)f2bf(a.w) << 16);
  r.z = (unsigned)f2bf(b.x) | ((unsigned)f2bf(b.y) << 16);
  r.w = (unsigned)f2bf(b.z) | ((unsigned)f2bf(b.w) << 16);
  return r;
}

DEV v8f wmma_bf16(v16bf a, v16bf b, v8f c) {
  // (neg_a, A, neg_b, B, c_mod, C, reuse_a, reuse_b)
  return __builtin_amdgcn_wmma_f32_16x16x32_bf16(false, a, false, b, (short)0, c,
                                                 false, false);
}

DEV void zero8(v8f acc[8]) {
  v8f z = {0.f, 0.f, 0.f, 0.f, 0.f, 0.f, 0.f, 0.f};
#pragma unroll
  for (int i = 0; i < 8; ++i) acc[i] = z;
}

// --- CDNA5 async global->LDS copy (ASYNCcnt path) ---------------------------
// Flat aperture rule: the LDS address is the low 32 bits of the generic address, so truncating a
// generic pointer into __shared__ yields the wave-relative LDS byte offset,
// which is exactly what VDST of global_load_async_to_lds expects.
DEV unsigned lds_off_of(const void* p) { return (unsigned)(uintptr_t)p; }

DEV void async_copy16(unsigned lds_byte_off, const void* gptr) {
  asm volatile("global_load_async_to_lds_b128 %0, %1, off"
               :
               : "v"(lds_byte_off), "v"((unsigned long long)(uintptr_t)gptr)
               : "memory");
}
DEV void wait_async0() {
  asm volatile("s_wait_asynccnt 0x0" ::: "memory");
}

// 16xK (LDS, row-major bf16) times Kx128 (global, stored output-major Wt[n][k])
// accumulated into 8 N-tiles of 16x16 f32.
// A-fragment (ISA 16-bit A layout): lane l -> row l&15; K = {8g..8g+7, 16+8g..23+8g}+32*kk
// B-fragment (ISA 16-bit B layout): lane l -> col l&15; K = {8g..8g+7, 16+8g..23+8g}+32*kk
template <int K>
DEV void gemm_tile(const unsigned short* X, const unsigned short* Wt, v8f acc[8]) {
  const int lane = threadIdx.x & 31;
  const int r = lane & 15;
  const int g = lane >> 4;
#pragma unroll
  for (int kk = 0; kk < K / 32; ++kk) {
    Frag a;
    const unsigned short* ap = X + r * K + kk * 32 + g * 8;
    a.q[0] = *(const uint4*)(ap);
    a.q[1] = *(const uint4*)(ap + 16);
#pragma unroll
    for (int nt = 0; nt < 8; ++nt) {
      Frag b;
      const unsigned short* bp = Wt + (size_t)(nt * 16 + r) * K + kk * 32 + g * 8;
      b.q[0] = *(const uint4*)(bp);
      b.q[1] = *(const uint4*)(bp + 16);
      acc[nt] = wmma_bf16(a.b, b.b, acc[nt]);
    }
  }
}

// ---------------------------------------------------------------------------
__global__ void k_zero(float* p, long n) {
  long i = (long)blockIdx.x * blockDim.x + threadIdx.x;
  long stride = (long)gridDim.x * blockDim.x;
  for (; i < n; i += stride) p[i] = 0.f;
}

// Build bf16 output-major weights + folded biases (gl == bg, so the 4th
// 128-row block of W1e/W1n collapses into a constant bias vector).
__global__ void k_prep(const float* W1e, const float* b1e, const float* W2e,
                       const float* W1n, const float* b1n, const float* W2n,
                       const float* W1g, const float* Wn, const float* We,
                       const float* bg,
                       unsigned short* W1e_t, unsigned short* W2e_t,
                       unsigned short* W1n_t, unsigned short* W2n_t,
                       unsigned short* W1g_t, unsigned short* Wn_t,
                       unsigned short* We_t, float* cb_e, float* cb_n) {
  const long TOTAL = 49152L * 3 + 16384L * 2 + 4096L * 2 + 256;
  long i = (long)blockIdx.x * blockDim.x + threadIdx.x;
  long stride = (long)gridDim.x * blockDim.x;
  for (; i < TOTAL; i += stride) {
    long idx = i;
    if (idx < 49152) { int n = idx / 384, k = idx % 384; W1e_t[idx] = f2bf(W1e[k * 128 + n]); continue; }
    idx -= 49152;
    if (idx < 16384) { int n = idx / 128, k = idx % 128; W2e_t[idx] = f2bf(W2e[k * 128 + n]); continue; }
    idx -= 16384;
    if (idx < 49152) { int n = idx / 384, k = idx % 384; W1n_t[idx] = f2bf(W1n[k * 128 + n]); continue; }
    idx -= 49152;
    if (idx < 16384) { int n = idx / 128, k = idx % 128; W2n_t[idx] = f2bf(W2n[k * 128 + n]); continue; }
    idx -= 16384;
    if (idx < 49152) { int n = idx / 384, k = idx % 384; W1g_t[idx] = f2bf(W1g[k * 128 + n]); continue; }
    idx -= 49152;
    if (idx < 4096) { int n = idx / 32, k = idx % 32; Wn_t[idx] = f2bf(Wn[k * 128 + n]); continue; }
    idx -= 4096;
    if (idx < 4096) { int n = idx / 32, k = idx % 32;
      We_t[idx] = (k < 16) ? f2bf(We[k * 128 + n]) : (unsigned short)0; continue; }
    idx -= 4096;
    if (idx < 128) { float s = b1e[idx];
      for (int k = 0; k < 128; ++k) s += bg[k] * W1e[(384 + k) * 128 + idx];
      cb_e[idx] = s; continue; }
    idx -= 128;
    { float s = b1n[idx];
      for (int k = 0; k < 128; ++k) s += bg[k] * W1n[(384 + k) * 128 + idx];
      cb_n[idx] = s; }
  }
}

// Embedding: out_bf16[rows][128] = bf16(X[rows][Kin] @ W + bias). Kin padded to 32
// (We_t has zeros for k>=16). One 16-row tile per wave, 4 waves/block.
__global__ __launch_bounds__(128) void k_embed(const float* X, const unsigned short* Wt,
                                               const float* bias, unsigned short* out_bf,
                                               int rows, int Kin) {
  int wv = threadIdx.x >> 5;
  int tile = blockIdx.x * 4 + wv;
  if (tile * 16 >= rows) return;
  int lane = threadIdx.x & 31, r = lane & 15, g = lane >> 4;
  int row = tile * 16 + r;
  Frag a;
  if (Kin == 32) {
    const float* ap = X + (size_t)row * 32 + g * 8;
    a.q[0] = pack8(ap);
    a.q[1] = pack8(ap + 16);
  } else {  // Kin == 16, high half of K is zero-padded
    const float* ap = X + (size_t)row * 16 + g * 8;
    a.q[0] = pack8(ap);
    a.q[1] = make_uint4(0, 0, 0, 0);
  }
  v8f z = {0.f, 0.f, 0.f, 0.f, 0.f, 0.f, 0.f, 0.f};
#pragma unroll
  for (int nt = 0; nt < 8; ++nt) {
    Frag b;
    const unsigned short* bp = Wt + (size_t)(nt * 16 + r) * 32 + g * 8;
    b.q[0] = *(const uint4*)(bp);
    b.q[1] = *(const uint4*)(bp + 16);
    v8f acc = wmma_bf16(a.b, b.b, z);
    int n = nt * 16 + r;
    float bs = bias[n];
#pragma unroll
    for (int v = 0; v < 8; ++v) {  // C layout: m = v + 8g, col = n
      int m = v + 8 * g;
      out_bf[(size_t)(tile * 16 + m) * 128 + n] = f2bf(acc[v] + bs);
    }
  }
}

// Edge MLP: per wave tile of 16 edges. X = [e | n[s] | n[r]] (16x384 bf16 in LDS,
// gathered via async global->LDS copies), h = relu(X@W1t + cb1) staged bf16 in
// LDS, e2 = h@W2t + b2 -> store + 3 scatter-adds.
__global__ __launch_bounds__(128) void k_edge_mlp(
    const unsigned short* e_bf, const unsigned short* n_bf, const int* senders,
    const int* receivers, const int* egidx, const unsigned short* W1t,
    const float* cb1, const unsigned short* W2t, const float* b2, float* e2_out,
    float* sent_agg, float* recv_agg, float* edge_agg, int E) {
  __shared__ unsigned short lds[4 * 8192];  // per wave: 16*384 X + 16*128 h
  int wv = threadIdx.x >> 5;
  int tile = blockIdx.x * 4 + wv;
  if (tile * 16 >= E) return;
  // Warm WGP$/L2 for the weight streams every wave re-reads.
  __builtin_prefetch(W1t + (size_t)threadIdx.x * 256, 0, 3);
  __builtin_prefetch(W1t + 32768 + (size_t)threadIdx.x * 128, 0, 3);
  __builtin_prefetch(W2t + (size_t)threadIdx.x * 128, 0, 3);
  int lane = threadIdx.x & 31, r = lane & 15, g = lane >> 4;
  unsigned short* lx = lds + wv * 8192;
  unsigned short* lh = lx + 6144;
  int e0 = tile * 16;
  // Gather X tile: 768 chunks of 8 bf16 (16B each) straight into LDS via the
  // ASYNCcnt path; the n-table (12.8MB) is L2-resident.
  unsigned lxo = lds_off_of(lx);
  for (int t = lane; t < 768; t += 32) {
    int row = t / 48, ch = t % 48, seg = ch >> 4, coff = (ch & 15) * 8;
    int erow = e0 + row;
    const unsigned short* src;
    if (seg == 0)      src = e_bf + (size_t)erow * 128 + coff;
    else if (seg == 1) src = n_bf + (size_t)senders[erow] * 128 + coff;
    else               src = n_bf + (size_t)receivers[erow] * 128 + coff;
    async_copy16(lxo + (unsigned)t * 16u, src);
  }
  wait_async0();
  v8f acc[8];
  zero8(acc);
  gemm_tile<384>(lx, W1t, acc);
#pragma unroll
  for (int nt = 0; nt < 8; ++nt) {
    int n = nt * 16 + r;
    float bs = cb1[n];
#pragma unroll
    for (int v = 0; v < 8; ++v) {
      float x = acc[nt][v] + bs;
      lh[(v + 8 * g) * 128 + n] = f2bf(x > 0.f ? x : 0.f);
    }
  }
  v8f acc2[8];
  zero8(acc2);
  gemm_tile<128>(lh, W2t, acc2);
  int si[8], ri[8], gi[8];
#pragma unroll
  for (int v = 0; v < 8; ++v) {
    int row = e0 + v + 8 * g;
    si[v] = senders[row];
    ri[v] = receivers[row];
    gi[v] = egidx[row];
  }
#pragma unroll
  for (int nt = 0; nt < 8; ++nt) {
    int n = nt * 16 + r;
    float bs = b2[n];
#pragma unroll
    for (int v = 0; v < 8; ++v) {
      float val = acc2[nt][v] + bs;
      int row = e0 + v + 8 * g;
      e2_out[(size_t)row * 128 + n] = val;
      atomicAdd(&sent_agg[(size_t)si[v] * 128 + n], val);
      atomicAdd(&recv_agg[(size_t)ri[v] * 128 + n], val);
      atomicAdd(&edge_agg[(size_t)gi[v] * 128 + n], val);
    }
  }
}

// Node MLP: X = [n | sent_agg | recv_agg]; the bf16 n-rows stream in via async
// copies, the f32 aggregates get converted in-register. Epilogue stores n2 and
// scatter-adds node_agg.
__global__ __launch_bounds__(128) void k_node_mlp(
    const unsigned short* n_bf, const float* sent_agg, const float* recv_agg,
    const int* ngidx, const unsigned short* W1t, const float* cb1,
    const unsigned short* W2t, const float* b2, float* n2_out, float* node_agg,
    int N) {
  __shared__ unsigned short lds[4 * 8192];
  int wv = threadIdx.x >> 5;
  int tile = blockIdx.x * 4 + wv;
  if (tile * 16 >= N) return;
  __builtin_prefetch(W1t + (size_t)threadIdx.x * 256, 0, 3);
  __builtin_prefetch(W1t + 32768 + (size_t)threadIdx.x * 128, 0, 3);
  __builtin_prefetch(W2t + (size_t)threadIdx.x * 128, 0, 3);
  int lane = threadIdx.x & 31, r = lane & 15, g = lane >> 4;
  unsigned short* lx = lds + wv * 8192;
  unsigned short* lh = lx + 6144;
  int n0 = tile * 16;
  unsigned lxo = lds_off_of(lx);
  for (int t = lane; t < 768; t += 32) {
    int row = t / 48, ch = t % 48, seg = ch >> 4, coff = (ch & 15) * 8;
    int nr = n0 + row;
    if (seg == 0) {
      async_copy16(lxo + (unsigned)t * 16u, n_bf + (size_t)nr * 128 + coff);
    } else {
      const float* s = (seg == 1 ? sent_agg : recv_agg) + (size_t)nr * 128 + coff;
      *(uint4*)(lx + row * 384 + ch * 8) = pack8(s);
    }
  }
  wait_async0();
  v8f acc[8];
  zero8(acc);
  gemm_tile<384>(lx, W1t, acc);
#pragma unroll
  for (int nt = 0; nt < 8; ++nt) {
    int n = nt * 16 + r;
    float bs = cb1[n];
#pragma unroll
    for (int v = 0; v < 8; ++v) {
      float x = acc[nt][v] + bs;
      lh[(v + 8 * g) * 128 + n] = f2bf(x > 0.f ? x : 0.f);
    }
  }
  v8f acc2[8];
  zero8(acc2);
  gemm_tile<128>(lh, W2t, acc2);
  int gi[8];
#pragma unroll
  for (int v = 0; v < 8; ++v) gi[v] = ngidx[n0 + v + 8 * g];
#pragma unroll
  for (int nt = 0; nt < 8; ++nt) {
    int n = nt * 16 + r;
    float bs = b2[n];
#pragma unroll
    for (int v = 0; v < 8; ++v) {
      float val = acc2[nt][v] + bs;
      int row = n0 + v + 8 * g;
      n2_out[(size_t)row * 128 + n] = val;
      atomicAdd(&node_agg[(size_t)gi[v] * 128 + n], val);
    }
  }
}

// Global MLP: G=128 rows. X = [node_agg | edge_agg | bg]; hidden via WMMA,
// tiny 128x3 output layer via VALU. 2 blocks x 4 waves x 16 rows.
__global__ __launch_bounds__(128) void k_global_mlp(
    const float* node_agg, const float* edge_agg, const float* bg,
    const unsigned short* W1gt, const float* b1g, const float* W2g,
    const float* b2g, float* g2_out) {
  __shared__ unsigned short lds[4 * 8192];
  int wv = threadIdx.x >> 5;
  int lane = threadIdx.x & 31, r = lane & 15, g = lane >> 4;
  unsigned short* lx = lds + wv * 8192;
  unsigned short* lh = lx + 6144;
  int g0 = blockIdx.x * 64 + wv * 16;
  for (int t = lane; t < 768; t += 32) {
    int row = t / 48, ch = t % 48, seg = ch >> 4, coff = (ch & 15) * 8;
    uint4 p;
    if (seg == 0)      p = pack8(node_agg + (size_t)(g0 + row) * 128 + coff);
    else if (seg == 1) p = pack8(edge_agg + (size_t)(g0 + row) * 128 + coff);
    else               p = pack8(bg + coff);  // gl row == bg for every graph
    *(uint4*)(lx + row * 384 + ch * 8) = p;
  }
  v8f acc[8];
  zero8(acc);
  gemm_tile<384>(lx, W1gt, acc);
#pragma unroll
  for (int nt = 0; nt < 8; ++nt) {
    int n = nt * 16 + r;
    float bs = b1g[n];
#pragma unroll
    for (int v = 0; v < 8; ++v) {
      float x = acc[nt][v] + bs;
      lh[(v + 8 * g) * 128 + n] = f2bf(x > 0.f ? x : 0.f);
    }
  }
  // out[16x3] = h[16x128] @ W2g[128x3] + b2g  (6K MACs per wave: VALU)
  for (int idx = lane; idx < 48; idx += 32) {
    int row = idx / 3, col = idx - row * 3;
    float s = b2g[col];
    for (int k = 0; k < 128; ++k) s += bf2f(lh[row * 128 + k]) * W2g[k * 3 + col];
    g2_out[(size_t)(g0 + row) * 3 + col] = s;
  }
}

// ---------------------------------------------------------------------------
extern "C" void kernel_launch(void* const* d_in, const int* in_sizes, int n_in,
                              void* d_out, int out_size, void* d_ws, size_t ws_size,
                              hipStream_t stream) {
  (void)n_in; (void)out_size; (void)ws_size;
  const float* nodes = (const float*)d_in[0];
  const float* edges = (const float*)d_in[1];
  const float* Wn  = (const float*)d_in[2];
  const float* bn  = (const float*)d_in[3];
  const float* We  = (const float*)d_in[4];
  const float* be  = (const float*)d_in[5];
  // d_in[6] (Wg) is multiplied by zeros in the reference -> unused
  const float* bg  = (const float*)d_in[7];
  const float* W1e = (const float*)d_in[8];
  const float* b1e = (const float*)d_in[9];
  const float* W2e = (const float*)d_in[10];
  const float* b2e = (const float*)d_in[11];
  const float* W1n = (const float*)d_in[12];
  const float* b1n = (const float*)d_in[13];
  const float* W2n = (const float*)d_in[14];
  const float* b2n = (const float*)d_in[15];
  const float* W1g = (const float*)d_in[16];
  const float* b1g = (const float*)d_in[17];
  const float* W2g = (const float*)d_in[18];
  const float* b2g = (const float*)d_in[19];
  const int* senders   = (const int*)d_in[20];
  const int* receivers = (const int*)d_in[21];
  const int* ngidx     = (const int*)d_in[22];
  const int* egidx     = (const int*)d_in[23];

  const int N = in_sizes[0] / 32;   // 50000
  const int E = in_sizes[1] / 16;   // 320000
  const int G = 128;

  // Workspace carve-up (256B aligned; agg buffers first and contiguous).
  char* ws = (char*)d_ws;
  size_t off = 0;
  auto carve = [&](size_t bytes) {
    size_t o = off;
    off += (bytes + 255) & ~(size_t)255;
    return o;
  };
  float* sent_agg = (float*)(ws + carve((size_t)N * 128 * 4));
  float* recv_agg = (float*)(ws + carve((size_t)N * 128 * 4));
  float* node_agg = (float*)(ws + carve((size_t)G * 128 * 4));
  float* edge_agg = (float*)(ws + carve((size_t)G * 128 * 4));
  float* cb_e     = (float*)(ws + carve(128 * 4));
  float* cb_n     = (float*)(ws + carve(128 * 4));
  unsigned short* n_bf  = (unsigned short*)(ws + carve((size_t)N * 128 * 2));
  unsigned short* e_bf  = (unsigned short*)(ws + carve((size_t)E * 128 * 2));
  unsigned short* W1e_t = (unsigned short*)(ws + carve(128 * 384 * 2));
  unsigned short* W2e_t = (unsigned short*)(ws + carve(128 * 128 * 2));
  unsigned short* W1n_t = (unsigned short*)(ws + carve(128 * 384 * 2));
  unsigned short* W2n_t = (unsigned short*)(ws + carve(128 * 128 * 2));
  unsigned short* W1g_t = (unsigned short*)(ws + carve(128 * 384 * 2));
  unsigned short* Wn_t  = (unsigned short*)(ws + carve(128 * 32 * 2));
  unsigned short* We_t  = (unsigned short*)(ws + carve(128 * 32 * 2));

  float* n2_out = (float*)d_out;
  float* e2_out = n2_out + (size_t)N * 128;
  float* g2_out = e2_out + (size_t)E * 128;

  // 1) zero aggregation buffers (contiguous at ws start)
  long zcount = ((long)2 * N + 2 * G) * 128;
  k_zero<<<2048, 256, 0, stream>>>(sent_agg, zcount);
  // 2) weight prep (bf16 output-major copies + folded biases)
  k_prep<<<768, 256, 0, stream>>>(W1e, b1e, W2e, W1n, b1n, W2n, W1g, Wn, We, bg,
                                  W1e_t, W2e_t, W1n_t, W2n_t, W1g_t, Wn_t, We_t,
                                  cb_e, cb_n);
  // 3) embeddings
  int ntiles = (N + 15) / 16;
  int etiles = (E + 15) / 16;
  k_embed<<<(ntiles + 3) / 4, 128, 0, stream>>>(nodes, Wn_t, bn, n_bf, N, 32);
  k_embed<<<(etiles + 3) / 4, 128, 0, stream>>>(edges, We_t, be, e_bf, E, 16);
  // 4) edge MLP + scatter
  k_edge_mlp<<<(etiles + 3) / 4, 128, 0, stream>>>(
      e_bf, n_bf, senders, receivers, egidx, W1e_t, cb_e, W2e_t, b2e, e2_out,
      sent_agg, recv_agg, edge_agg, E);
  // 5) node MLP + scatter
  k_node_mlp<<<(ntiles + 3) / 4, 128, 0, stream>>>(
      n_bf, sent_agg, recv_agg, ngidx, W1n_t, cb_n, W2n_t, b2n, n2_out,
      node_agg, N);
  // 6) global MLP
  k_global_mlp<<<2, 128, 0, stream>>>(node_agg, edge_agg, bg, W1g_t, b1g, W2g,
                                      b2g, g2_out);
}
